// WindowedAttention_62998580298273
// MI455X (gfx1250) — matrix operand order repaired and checked by reading, not compiled, not run on hardware
//
#include <hip/hip_runtime.h>

// ---------------------------------------------------------------------------
// Windowed attention (Swin-style) for MI455X / gfx1250.
// All matmuls in bf16 via v_wmma_f32_16x16x32_bf16 (f32 accumulate).
// GEMMs: 128x128x32 tiles, double-buffered LDS, b128 staging both operands.
// ---------------------------------------------------------------------------

typedef __attribute__((ext_vector_type(16))) __bf16 v16bf;
typedef __attribute__((ext_vector_type(8)))  float  v8f;

#define WIN   14
#define NH    12
#define HD    64
#define CDIM  768
#define IMH   80
#define IMW   80
#define BATCH 8
#define NTOK  (IMH * IMW)      // 6400
#define NWIN  6                // windows per side (84/14)
#define LWIN  (NWIN * NWIN)    // 36
#define NW    (WIN * WIN)      // 196 tokens per window
#define NWPAD 224              // padded to 7 chunks of 32 for PV contraction
#define NKT   13               // 13 key tiles of 16 (covers 208 >= 196)
#define AWAVES 7               // attention waves per block (224 threads)

// ---------------- WMMA fragment helpers (wave32, 16x16x32 bf16) ------------
// A (16x32): lane l: row = l&15; k = (l>>4)*16 + j
// B (32x16): lane l: col = l&15; k = (l>>4)*16 + j
// C (16x16 f32): vgpr r: row = (l>>4)*8 + r; col = l&15

__device__ __forceinline__ v16bf frag_a_from(const __bf16* row_base, int hi) {
  v16bf a;
  const __bf16* p0 = row_base + hi * 16;
  const __bf16* p1 = p0 + 8;
#pragma unroll
  for (int j = 0; j < 8; ++j) { a[j] = p0[j]; a[j + 8] = p1[j]; }
  return a;
}

__device__ __forceinline__ v16bf frag_b_from(const __bf16* col_base, int hi) {
  v16bf b;
  const __bf16* p = col_base + hi * 16;
#pragma unroll
  for (int j = 0; j < 16; ++j) b[j] = p[j];
  return b;
}

__device__ __forceinline__ v8f wmma_bf16(v16bf a, v16bf b, v8f c) {
  return __builtin_amdgcn_wmma_f32_16x16x32_bf16(
      /*neg_a=*/false, a, /*neg_b=*/false, b,
      /*c_mod=*/(short)0, c, /*reuse_a=*/false, /*reuse_b=*/false);
}

// ---------------- f32 -> bf16 conversion ----------------------------------
__global__ void cvt_f32_bf16(const float* __restrict__ in,
                             __bf16* __restrict__ out, size_t n) {
  size_t i = (size_t)blockIdx.x * blockDim.x + threadIdx.x;
  size_t stride = (size_t)gridDim.x * blockDim.x;
  for (; i < n; i += stride) out[i] = (__bf16)in[i];
}

// ---------------- Tiled bf16 GEMM: C[M,N] = A[M,K] * B[K,N] ----------------
// 256 threads = 8 waves. Block tile 128x128, K step 32, double-buffered LDS.
// Wave (wm in 0..1, wn in 0..3) computes a 64x32 region = 4x2 WMMA tiles.
// A staged [row][k] (b128 stores), B staged NATURAL [k][n] (b128 stores);
// B-fragment column reads are lane-contiguous (2 lanes/dword -> merge).
// OUT_BF16=1: store bf16 (no bias). OUT_BF16=0: store f32 + bias.
template <int OUT_BF16>
__global__ __launch_bounds__(256) void gemm_bf16_kernel(
    const __bf16* __restrict__ A, const __bf16* __restrict__ Bm,
    void* __restrict__ Cout, const float* __restrict__ bias,
    int M, int N, int K) {
  __shared__ __align__(16) __bf16 As[2][128][32];
  __shared__ __align__(16) __bf16 Bs[2][32][128];

  const int tid  = threadIdx.x;
  const int lane = tid & 31, wave = tid >> 5;
  const int l16  = lane & 15, hi = lane >> 4;
  const int m0 = blockIdx.y * 128, n0 = blockIdx.x * 128;
  const int wm = wave & 1, wn = wave >> 1;

  // staging assignments (each thread moves 32B of A and 32B of B per step)
  const int arow = tid >> 1, akc = (tid & 1) * 16;
  const int bkr  = tid >> 3, bnc = (tid & 7) * 16;
  const __bf16* aptr = A + (size_t)(m0 + arow) * K + akc;
  const __bf16* bptr = Bm + (size_t)bkr * N + n0 + bnc;

  v8f acc[4][2];
#pragma unroll
  for (int i = 0; i < 4; ++i)
#pragma unroll
    for (int j = 0; j < 2; ++j) acc[i][j] = (v8f)0.0f;

  uint4 ar0, ar1, br0, br1;
  // prologue: stage tile 0 into buffer 0
  ar0 = ((const uint4*)aptr)[0];
  ar1 = ((const uint4*)aptr)[1];
  br0 = ((const uint4*)bptr)[0];
  br1 = ((const uint4*)bptr)[1];
  ((uint4*)&As[0][arow][akc])[0] = ar0;
  ((uint4*)&As[0][arow][akc])[1] = ar1;
  ((uint4*)&Bs[0][bkr][bnc])[0] = br0;
  ((uint4*)&Bs[0][bkr][bnc])[1] = br1;

  const int KS = K / 32;
  for (int ks = 0; ks < KS; ++ks) {
    __syncthreads();
    const int cur = ks & 1, nxt = cur ^ 1;
    const bool more = (ks + 1) < KS;

    if (more) {  // issue next tile's global loads; they complete under WMMAs
      const __bf16* ap = aptr + (ks + 1) * 32;
      const __bf16* bp = bptr + (size_t)(ks + 1) * 32 * N;
      ar0 = ((const uint4*)ap)[0];
      ar1 = ((const uint4*)ap)[1];
      br0 = ((const uint4*)bp)[0];
      br1 = ((const uint4*)bp)[1];
      if (ks + 2 < KS) {  // cache hint for the tile after (global_prefetch_b8)
        __builtin_prefetch(aptr + (ks + 2) * 32, 0, 3);
        __builtin_prefetch(bptr + (size_t)(ks + 2) * 32 * N, 0, 3);
      }
    }

    // compute from current buffer
    v16bf af[4], bfr[2];
#pragma unroll
    for (int i = 0; i < 4; ++i)
      af[i] = frag_a_from(&As[cur][wm * 64 + i * 16 + l16][0], hi);
#pragma unroll
    for (int j = 0; j < 2; ++j) {
      const int n = wn * 32 + j * 16 + l16;
#pragma unroll
      for (int jj = 0; jj < 16; ++jj) bfr[j][jj] = Bs[cur][hi * 16 + jj][n];
    }
#pragma unroll
    for (int i = 0; i < 4; ++i)
#pragma unroll
      for (int j = 0; j < 2; ++j)
        acc[i][j] = wmma_bf16(af[i], bfr[j], acc[i][j]);

    if (more) {  // park next tile into the other buffer
      ((uint4*)&As[nxt][arow][akc])[0] = ar0;
      ((uint4*)&As[nxt][arow][akc])[1] = ar1;
      ((uint4*)&Bs[nxt][bkr][bnc])[0] = br0;
      ((uint4*)&Bs[nxt][bkr][bnc])[1] = br1;
    }
  }

  __bf16* cb = (__bf16*)Cout;
  float*  cf = (float*)Cout;
#pragma unroll
  for (int i = 0; i < 4; ++i) {
#pragma unroll
    for (int j = 0; j < 2; ++j) {
      int col = n0 + wn * 32 + j * 16 + l16;
      size_t base = (size_t)(m0 + wm * 64 + i * 16 + hi * 8) * N + col;
#pragma unroll
      for (int r = 0; r < 8; ++r) {
        float v = acc[i][j][r];
        size_t idx = base + (size_t)r * N;
        if constexpr (OUT_BF16) {
          cb[idx] = (__bf16)v;
        } else {
          cf[idx] = v + bias[col];
        }
      }
    }
  }
}

// ---------------- Windowed attention ---------------------------------------
// One block per (batch, window, head); 224 threads = 7 waves, 13 query strips
// split 2/2/2/2/2/2/1. qkv is bf16 [B, 6400, 3*768] ([3][nh][hd] innermost).
// Output scattered directly into merged [B, 6400, 768] (non-overlapping fold).
__global__ __launch_bounds__(224) void win_attn_kernel(
    const __bf16* __restrict__ qkv, __bf16* __restrict__ out) {
  extern __shared__ __align__(16) char smem_raw[];
  __bf16* kbuf  = (__bf16*)smem_raw;        // [NWPAD][HD]   keys (row = key)
  __bf16* vbufT = kbuf + NWPAD * HD;        // [HD][NWPAD]   V transposed
  __bf16* pbuf  = vbufT + HD * NWPAD;       // [AWAVES][16][NWPAD] P slabs

  const int tid  = threadIdx.x;
  const int lane = tid & 31, wave = tid >> 5;
  const int l16  = lane & 15, hi = lane >> 4;

  int blk = blockIdx.x;
  const int h = blk % NH;   blk /= NH;
  const int l = blk % LWIN; blk /= LWIN;
  const int b = blk;
  const int wh = l / NWIN, wwi = l % NWIN;

  // ---- stage K and V^T into LDS; image-pad tokens and rows>=196 are zero
  for (int task = tid; task < NWPAD * 8; task += 224) {
    const int t = task >> 3, dc = (task & 7) * 8;
    int rr = t / WIN, cc = t % WIN;
    int ih = wh * WIN + rr, iw = wwi * WIN + cc;
    bool valid = (t < NW) && (ih < IMH) && (iw < IMW);
    __bf16 kv[8], vv[8];
    if (valid) {
      size_t base = ((size_t)(b * NTOK + ih * IMW + iw)) * (3 * CDIM) + h * HD + dc;
      const __bf16* kp = qkv + base + CDIM;
      const __bf16* vp = qkv + base + 2 * CDIM;
#pragma unroll
      for (int j = 0; j < 8; ++j) { kv[j] = kp[j]; vv[j] = vp[j]; }
    } else {
#pragma unroll
      for (int j = 0; j < 8; ++j) { kv[j] = (__bf16)0.0f; vv[j] = (__bf16)0.0f; }
    }
#pragma unroll
    for (int j = 0; j < 8; ++j) {
      kbuf[t * HD + dc + j]        = kv[j];
      vbufT[(dc + j) * NWPAD + t]  = vv[j];
    }
  }
  __syncthreads();

  const float scale = 0.125f;  // hd^-0.5 = 1/8

  for (int strip = wave; strip < NKT; strip += AWAVES) {
    // ---- Q fragments directly from global (bf16, contiguous per lane)
    int t  = strip * 16 + l16;
    int rr = t / WIN, cc = t % WIN;
    int ih = wh * WIN + rr, iw = wwi * WIN + cc;
    bool qvalid = (t < NW) && (ih < IMH) && (iw < IMW);
    v16bf qa[2];
    if (qvalid) {
      const __bf16* qp =
          qkv + ((size_t)(b * NTOK + ih * IMW + iw)) * (3 * CDIM) + h * HD;
#pragma unroll
      for (int kc = 0; kc < 2; ++kc) {
        const __bf16* p0 = qp + kc * 32 + hi * 16;
        const __bf16* p1 = p0 + 8;
#pragma unroll
        for (int j = 0; j < 8; ++j) { qa[kc][j] = p0[j]; qa[kc][j + 8] = p1[j]; }
      }
    } else {
#pragma unroll
      for (int kc = 0; kc < 2; ++kc)
#pragma unroll
        for (int j = 0; j < 16; ++j) qa[kc][j] = (__bf16)0.0f;
    }

    // ---- S = Q K^T (per key tile: 2 WMMAs over hd=64)
    v8f s[NKT];
#pragma unroll
    for (int kt = 0; kt < NKT; ++kt) {
      v8f c = (v8f)0.0f;
      v16bf b0 = frag_b_from(&kbuf[(kt * 16 + l16) * HD + 0], hi);
      v16bf b1 = frag_b_from(&kbuf[(kt * 16 + l16) * HD + 32], hi);
      c = wmma_bf16(qa[0], b0, c);
      c = wmma_bf16(qa[1], b1, c);
      s[kt] = c;
    }

    // ---- softmax over 196 keys (cols >= 196 masked; image-pad keys are
    // genuine zero tokens, matching the reference which does NOT mask them)
    float rmax[8], rsum[8];
#pragma unroll
    for (int r = 0; r < 8; ++r) rmax[r] = -3.0e38f;
#pragma unroll
    for (int kt = 0; kt < NKT; ++kt) {
      bool colok = (kt * 16 + l16) < NW;
#pragma unroll
      for (int r = 0; r < 8; ++r) {
        float v = s[kt][r] * scale;
        if (colok && v > rmax[r]) rmax[r] = v;
      }
    }
#pragma unroll
    for (int r = 0; r < 8; ++r) {
#pragma unroll
      for (int off = 1; off < 16; off <<= 1)
        rmax[r] = fmaxf(rmax[r], __shfl_xor(rmax[r], off, 32));
    }
#pragma unroll
    for (int r = 0; r < 8; ++r) rsum[r] = 0.0f;

    __bf16* pb = pbuf + wave * 16 * NWPAD;  // wave-private P slab [16][NWPAD]
#pragma unroll
    for (int kt = 0; kt < NKT; ++kt) {
      int col = kt * 16 + l16;
      bool colok = col < NW;
#pragma unroll
      for (int r = 0; r < 8; ++r) {
        float p = colok ? __expf(s[kt][r] * scale - rmax[r]) : 0.0f;
        rsum[r] += p;
        pb[(hi * 8 + r) * NWPAD + col] = (__bf16)p;
      }
    }
#pragma unroll
    for (int r = 0; r < 8; ++r)  // zero tail cols 208..223
      pb[(hi * 8 + r) * NWPAD + 208 + l16] = (__bf16)0.0f;
#pragma unroll
    for (int r = 0; r < 8; ++r) {
#pragma unroll
      for (int off = 1; off < 16; off <<= 1)
        rsum[r] += __shfl_xor(rsum[r], off, 32);
    }
    // wave-local LDS store->load ordering for the P slab
    asm volatile("s_wait_dscnt 0" ::: "memory");

    // ---- O = P V (7 K-chunks of 32 over padded 224 keys; 4 hd tiles)
    v8f o[4];
#pragma unroll
    for (int nt = 0; nt < 4; ++nt) o[nt] = (v8f)0.0f;
#pragma unroll
    for (int kc = 0; kc < 7; ++kc) {
      v16bf pa = frag_a_from(&pb[l16 * NWPAD + kc * 32], hi);
#pragma unroll
      for (int nt = 0; nt < 4; ++nt) {
        v16bf vb = frag_b_from(&vbufT[(nt * 16 + l16) * NWPAD + kc * 32], hi);
        o[nt] = wmma_bf16(pa, vb, o[nt]);
      }
    }

    // ---- normalize and scatter into merged [B,6400,768] layout
    int trow = strip * 16 + hi * 8;
#pragma unroll
    for (int r = 0; r < 8; ++r) {
      int tok = trow + r;
      int rr2 = tok / WIN, cc2 = tok % WIN;
      int ih2 = wh * WIN + rr2, iw2 = wwi * WIN + cc2;
      if (tok < NW && ih2 < IMH && iw2 < IMW) {
        float inv = 1.0f / rsum[r];
        size_t base =
            ((size_t)(b * NTOK + ih2 * IMW + iw2)) * CDIM + h * HD + l16;
#pragma unroll
        for (int nt = 0; nt < 4; ++nt)
          out[base + nt * 16] = (__bf16)(o[nt][r] * inv);
      }
    }
  }
}

// ---------------------------------------------------------------------------
extern "C" void kernel_launch(void* const* d_in, const int* in_sizes, int n_in,
                              void* d_out, int out_size, void* d_ws,
                              size_t ws_size, hipStream_t stream) {
  (void)in_sizes; (void)n_in; (void)out_size; (void)ws_size;

  const float* x     = (const float*)d_in[0];
  const float* Wqkv  = (const float*)d_in[1];
  const float* Wproj = (const float*)d_in[2];
  const float* bproj = (const float*)d_in[3];
  // d_in[4], d_in[5] are H=80, W=80 (compile-time constants here)

  const size_t nX   = (size_t)BATCH * NTOK * CDIM;      // 39,321,600
  const size_t nWq  = (size_t)CDIM * 3 * CDIM;          //  1,769,472
  const size_t nWp  = (size_t)CDIM * CDIM;              //    589,824
  const size_t nQKV = (size_t)BATCH * NTOK * 3 * CDIM;  // 117,964,800

  auto align256 = [](size_t v) { return (v + 255) & ~(size_t)255; };
  char* ws = (char*)d_ws;
  __bf16* x_bf   = (__bf16*)ws; ws += align256(nX * 2);
  __bf16* wq_bf  = (__bf16*)ws; ws += align256(nWq * 2);
  __bf16* wp_bf  = (__bf16*)ws; ws += align256(nWp * 2);
  __bf16* qkv_bf = (__bf16*)ws; ws += align256(nQKV * 2);
  __bf16* ao_bf  = (__bf16*)ws; ws += align256(nX * 2);

  cvt_f32_bf16<<<4096, 256, 0, stream>>>(x, x_bf, nX);
  cvt_f32_bf16<<<1024, 256, 0, stream>>>(Wqkv, wq_bf, nWq);
  cvt_f32_bf16<<<512, 256, 0, stream>>>(Wproj, wp_bf, nWp);

  // QKV projection: [51200,768] x [768,2304] -> bf16
  dim3 g1(3 * CDIM / 128, (BATCH * NTOK) / 128);  // (18, 400)
  gemm_bf16_kernel<1><<<g1, 256, 0, stream>>>(x_bf, wq_bf, qkv_bf, nullptr,
                                              BATCH * NTOK, 3 * CDIM, CDIM);

  // Windowed attention: one block per (b, window, head); ~105 KB dynamic LDS
  const size_t attn_lds =
      (size_t)(NWPAD * HD + HD * NWPAD + AWAVES * 16 * NWPAD) * 2;
  win_attn_kernel<<<BATCH * LWIN * NH, 224, attn_lds, stream>>>(qkv_bf, ao_bf);

  // Output projection: [51200,768] x [768,768] + bias -> f32 d_out
  dim3 g2(CDIM / 128, (BATCH * NTOK) / 128);  // (6, 400)
  gemm_bf16_kernel<0><<<g2, 256, 0, stream>>>(ao_bf, wp_bf, d_out, bproj,
                                              BATCH * NTOK, CDIM, CDIM);
}
